// ContactMapGeneration_3659312136828
// MI455X (gfx1250) — compile-verified, matrix operand order checked by reading, not run on hardware
//
#include <hip/hip_runtime.h>

typedef __attribute__((ext_vector_type(16))) _Float16 v16h;
typedef __attribute__((ext_vector_type(8)))  _Float16 v8h;
typedef __attribute__((ext_vector_type(8)))  float    v8f;

#define BATCH  16
#define NPTS   8192
#define NPT    2048
#define MROWS  (BATCH * NPT)   // 32768 rows total

// ---------------------------------------------------------------------------
// 1) centroid + s_obj = max ||p - centroid|| per batch
// ---------------------------------------------------------------------------
__global__ void k_scale(const float* __restrict__ mesh, float* __restrict__ s_obj) {
  int b = blockIdx.x, t = threadIdx.x;
  __shared__ float sx[256], sy[256], sz[256];
  __shared__ float cx, cy, cz;
  const float* mb = mesh + (size_t)b * NPTS * 3;
  float ax = 0.f, ay = 0.f, az = 0.f;
  for (int p = t; p < NPTS; p += 256) { ax += mb[p*3+0]; ay += mb[p*3+1]; az += mb[p*3+2]; }
  sx[t] = ax; sy[t] = ay; sz[t] = az; __syncthreads();
  for (int s = 128; s > 0; s >>= 1) {
    if (t < s) { sx[t] += sx[t+s]; sy[t] += sy[t+s]; sz[t] += sz[t+s]; }
    __syncthreads();
  }
  if (t == 0) { cx = sx[0] / NPTS; cy = sy[0] / NPTS; cz = sz[0] / NPTS; }
  __syncthreads();
  float mx = 0.f;
  for (int p = t; p < NPTS; p += 256) {
    float dx = mb[p*3+0]-cx, dy = mb[p*3+1]-cy, dz = mb[p*3+2]-cz;
    mx = fmaxf(mx, sqrtf(dx*dx + dy*dy + dz*dz));
  }
  sx[t] = mx; __syncthreads();
  for (int s = 128; s > 0; s >>= 1) {
    if (t < s) sx[t] = fmaxf(sx[t], sx[t+s]);
    __syncthreads();
  }
  if (t == 0) s_obj[b] = sx[0];
}

// ---------------------------------------------------------------------------
// 2) Farthest point sampling. One 1024-thread workgroup per batch.
//    xyz in LDS (broadcast reads), running distance in REGISTERS (8/thread),
//    argmax via wave32 shfl_xor reduce -> only 2 barriers per iteration.
//    Tie-break to lowest index == jnp.argmax first-occurrence semantics.
// ---------------------------------------------------------------------------
__global__ void __launch_bounds__(1024) k_fps(const float* __restrict__ mesh,
                                              const int* __restrict__ fps_init,
                                              int* __restrict__ idx) {
  extern __shared__ float smem[];
  float* lx  = smem;
  float* ly  = lx + NPTS;
  float* lz  = ly + NPTS;
  float* rv  = lz + NPTS;        // [32] per-wave best value
  int*   ri  = (int*)(rv + 32);  // [32] per-wave best index
  int*   sfar = ri + 32;         // [1]

  int b = blockIdx.x, t = threadIdx.x;
  int lane = t & 31, wave = t >> 5;
  const float* mb = mesh + (size_t)b * NPTS * 3;
  for (int p = t; p < NPTS; p += 1024) {
    lx[p] = mb[p*3+0]; ly[p] = mb[p*3+1]; lz[p] = mb[p*3+2];
  }
  float dreg[8];
#pragma unroll
  for (int j = 0; j < 8; ++j) dreg[j] = 1e10f;
  __syncthreads();

  int far = fps_init[b];
  for (int i = 0; i < NPT; ++i) {
    if (t == 0) idx[b * NPT + i] = far;            // record BEFORE update (scan)
    float fx = lx[far], fy = ly[far], fz = lz[far];
    float bv = -1.f; int bi = 0x7fffffff;
#pragma unroll
    for (int j = 0; j < 8; ++j) {
      int p = t + j * 1024;                        // ascending p within thread
      float dx = lx[p]-fx, dy = ly[p]-fy, dz = lz[p]-fz;
      float d  = dx*dx + dy*dy + dz*dz;
      float nd = fminf(dreg[j], d);
      dreg[j] = nd;
      if (nd > bv) { bv = nd; bi = p; }
    }
    // wave32 reduce (no barriers)
#pragma unroll
    for (int off = 16; off > 0; off >>= 1) {
      float ov = __shfl_xor(bv, off, 32);
      int   oi = __shfl_xor(bi, off, 32);
      if (ov > bv || (ov == bv && oi < bi)) { bv = ov; bi = oi; }
    }
    if (lane == 0) { rv[wave] = bv; ri[wave] = bi; }
    __syncthreads();
    // cross-wave reduce in wave 0
    if (wave == 0) {
      float wv = rv[lane]; int wi = ri[lane];
#pragma unroll
      for (int off = 16; off > 0; off >>= 1) {
        float ov = __shfl_xor(wv, off, 32);
        int   oi = __shfl_xor(wi, off, 32);
        if (ov > wv || (ov == wv && oi < wi)) { wv = ov; wi = oi; }
      }
      if (lane == 0) *sfar = wi;
    }
    __syncthreads();
    far = *sfar;
  }
}

// ---------------------------------------------------------------------------
// 3) gather sampled points, normalize, build pcc = [cm, xyz/s], samp_cm out
// ---------------------------------------------------------------------------
__global__ void k_gather(const float* __restrict__ mesh, const float* __restrict__ cmap,
                         const int* __restrict__ idx, const float* __restrict__ s_obj,
                         float* __restrict__ pc_norm, float* __restrict__ pcc,
                         float* __restrict__ samp_out) {
  int i = blockIdx.x * 256 + threadIdx.x;
  if (i >= MROWS) return;
  int b = i >> 11;
  int id = idx[i];
  float inv = 1.f / s_obj[b];
  const float* mp = mesh + ((size_t)b * NPTS + id) * 3;
  float x = mp[0]*inv, y = mp[1]*inv, z = mp[2]*inv;
  float cm = cmap[(size_t)b * NPTS + id];
  pc_norm[i*3+0] = x; pc_norm[i*3+1] = y; pc_norm[i*3+2] = z;
  pcc[i*4+0] = cm; pcc[i*4+1] = x; pcc[i*4+2] = y; pcc[i*4+3] = z;
  samp_out[i] = cm;
}

// ---------------------------------------------------------------------------
// 4) first layer (din=3 or 4 -> 64) scalar, relu, f16 output
// ---------------------------------------------------------------------------
__global__ void k_layer1(const float* __restrict__ X, int din, const float* __restrict__ W,
                         const float* __restrict__ bias, _Float16* __restrict__ Y) {
  __shared__ float w[4*64 + 64];
  int t = threadIdx.x;
  if (t < din * 64) w[t] = W[t];
  if (t < 64)       w[din*64 + t] = bias[t];
  __syncthreads();
  int col = t & 63, r = t >> 6;
  int m = blockIdx.x * 4 + r;
  const float* x = X + (size_t)m * din;
  float acc = w[din*64 + col];
  for (int k = 0; k < din; ++k) acc += x[k] * w[k*64 + col];
  Y[(size_t)m * 64 + col] = (_Float16)fmaxf(acc, 0.f);
}

// ---------------------------------------------------------------------------
// 5) pack f32 weights [K,N] (row offset for dec1 slice) into wave32 WMMA
//    B-fragment layout: frag(nt,kt): 32 lanes x 16 f16, lane=column n%16,
//    K packed per ISA 7.12.2 (halves split across lane groups).
// ---------------------------------------------------------------------------
__global__ void k_pack(const float* __restrict__ W, int K, int N, int rowOff,
                       _Float16* __restrict__ out) {
  int f = blockIdx.x * 256 + threadIdx.x;
  if (f >= K * N) return;
  int KT = K >> 5;
  int e = f & 15, lane = (f >> 4) & 31, rest = f >> 9;
  int kt = rest % KT, nt = rest / KT;
  int half = e >> 3, off = e & 7, hi = lane >> 4;
  int k = kt*32 + half*16 + hi*8 + off;
  int n = nt*16 + (lane & 15);
  out[f] = (_Float16)W[(size_t)(k + rowOff) * N + n];
}

// ---------------------------------------------------------------------------
// 6) WMMA GEMM: C[M,N] = act(A[M,K] @ W [+ bias] [+ rowBias[batch]])
//    One wave per 16-row strip; A fragments in registers across all N-tiles.
//    Per-row output pointers precomputed (column advances 16/tile).
//    ACT: 0 none, 1 relu, 2 leaky(0.02)
// ---------------------------------------------------------------------------
template<int KT, int ACT, bool ROWB, bool HASB>
__global__ void __launch_bounds__(256) k_wmma(const _Float16* __restrict__ A,
                                              const _Float16* __restrict__ Wp,
                                              const float* __restrict__ bias,
                                              const float* __restrict__ rowBias,
                                              _Float16* __restrict__ C, int N) {
  const int K = KT * 32;
  int lane = threadIdx.x & 31;
  int mt = blockIdx.x * 8 + (threadIdx.x >> 5);
  int m0 = mt << 4;
  int hi = lane >> 4;
  int mrow = m0 + (lane & 15);
  const _Float16* arow = A + (size_t)mrow * K;

  v16h a[KT];
#pragma unroll
  for (int kt = 0; kt < KT; ++kt) {
    v16h tv;
    *(v8h*)&tv       = *(const v8h*)(arow + kt*32 + hi*8);
    *((v8h*)&tv + 1) = *(const v8h*)(arow + kt*32 + 16 + hi*8);
    a[kt] = tv;
  }

  _Float16* crow[8];
#pragma unroll
  for (int r = 0; r < 8; ++r)
    crow[r] = C + (size_t)(m0 + hi*8 + r) * N + (lane & 15);

  int NT = N >> 4;
  int batch = m0 >> 11;                      // 2048 rows per batch
  const float* rb = ROWB ? (rowBias + batch * N + (lane & 15)) : nullptr;
  const float* bb = HASB ? (bias + (lane & 15)) : nullptr;

  for (int nt = 0; nt < NT; ++nt) {
    const _Float16* wp = Wp + (((size_t)nt * KT) << 9) + (lane << 4);
    if (nt + 1 < NT) __builtin_prefetch(wp + (KT << 9), 0, 1);
    v8f acc = {};
#pragma unroll
    for (int kt = 0; kt < KT; ++kt) {
      v16h bfr = *(const v16h*)(wp + (kt << 9));
      acc = __builtin_amdgcn_wmma_f32_16x16x32_f16(false, a[kt], false, bfr,
                                                   (short)0, acc, false, false);
    }
    int ncol = nt << 4;
    float add = 0.f;
    if (HASB) add += bb[ncol];
    if (ROWB) add += rb[ncol];
#pragma unroll
    for (int r = 0; r < 8; ++r) {
      float v = acc[r] + add;
      if (ACT == 1)      v = fmaxf(v, 0.f);
      else if (ACT == 2) v = v > 0.f ? v : 0.02f * v;
      crow[r][ncol] = (_Float16)v;
    }
  }
}

// ---------------------------------------------------------------------------
// 7) maxpool over P points per batch (h3 is relu'd, so >= 0)
// ---------------------------------------------------------------------------
__global__ void k_maxpool(const _Float16* __restrict__ H, int Cc, float* __restrict__ out) {
  int col = blockIdx.x * 256 + threadIdx.x;
  int b = blockIdx.y;
  if (col >= Cc) return;
  const _Float16* hp = H + (size_t)b * NPT * Cc + col;
  float m = -1e30f;
  for (int p = 0; p < NPT; ++p) m = fmaxf(m, (float)hp[(size_t)p * Cc]);
  out[b * Cc + col] = m;
}

// ---------------------------------------------------------------------------
// 8) text feature: mean of 77 embedding rows -> [B,512]
// ---------------------------------------------------------------------------
__global__ void k_text(const float* __restrict__ emb, const int* __restrict__ toks,
                       float* __restrict__ out) {
  int b = blockIdx.x, c = threadIdx.x;       // 512 threads
  float s = 0.f;
  for (int j = 0; j < 77; ++j) s += emb[(size_t)toks[b*77+j] * 512 + c];
  out[b * 512 + c] = s * (1.f / 77.f);
}

// ---------------------------------------------------------------------------
// 9) small dense f32 (rows = 16 batches only)
// ---------------------------------------------------------------------------
__global__ void k_dense(const float* __restrict__ X, const float* __restrict__ W,
                        const float* __restrict__ bias, float* __restrict__ Y,
                        int din, int dout, int act) {
  int b = blockIdx.x;
  const float* x = X + (size_t)b * din;
  for (int j = threadIdx.x; j < dout; j += blockDim.x) {
    float acc = bias[j];
    for (int k = 0; k < din; ++k) acc += x[k] * W[(size_t)k * dout + j];
    if (act == 1) acc = fmaxf(acc, 0.f);
    Y[(size_t)b * dout + j] = acc;
  }
}

// ---------------------------------------------------------------------------
// 10) reparameterize: mu/log_var outputs + contact_vec
// ---------------------------------------------------------------------------
__global__ void k_reparam(const float* __restrict__ h, const float* __restrict__ eps,
                          float* __restrict__ mu_out, float* __restrict__ lv_out,
                          float* __restrict__ cv) {
  int b = blockIdx.x, j = threadIdx.x;       // 64 threads
  float mu = h[b*128 + j], lv = h[b*128 + 64 + j];
  mu_out[b*64 + j] = mu;
  lv_out[b*64 + j] = lv;
  cv[b*64 + j] = mu + eps[b*64 + j] * expf(0.5f * lv);
}

// ---------------------------------------------------------------------------
// 11) per-batch decoder-1 bias: b1 + gf@W1[0:1024] + tf@W1[1088:1600]
//     + s_obj*W1[1600] + cv@W1[1601:1665]   (broadcast features folded out)
// ---------------------------------------------------------------------------
__global__ void k_pbias(const float* __restrict__ gf, const float* __restrict__ tf,
                        const float* __restrict__ s_obj, const float* __restrict__ cv,
                        const float* __restrict__ W1, const float* __restrict__ b1,
                        float* __restrict__ pb) {
  int b = blockIdx.x, j = threadIdx.x;       // 512 threads
  float acc = b1[j];
  for (int k = 0; k < 1024; ++k) acc += gf[b*1024 + k] * W1[(size_t)k * 512 + j];
  for (int k = 0; k < 512;  ++k) acc += tf[b*512  + k] * W1[(size_t)(1088 + k) * 512 + j];
  acc += s_obj[b] * W1[(size_t)1600 * 512 + j];
  for (int k = 0; k < 64;   ++k) acc += cv[b*64   + k] * W1[(size_t)(1601 + k) * 512 + j];
  pb[b*512 + j] = acc;
}

// ---------------------------------------------------------------------------
// 12) decoder layer 4: [32768,128] @ [128,1] + b  (no activation)
// ---------------------------------------------------------------------------
__global__ void k_dec4(const _Float16* __restrict__ D3, const float* __restrict__ w4,
                       const float* __restrict__ b4, float* __restrict__ out) {
  int m = blockIdx.x * 256 + threadIdx.x;
  if (m >= MROWS) return;
  const _Float16* x = D3 + (size_t)m * 128;
  float acc = b4[0];
  for (int k = 0; k < 128; ++k) acc += (float)x[k] * w4[k];
  out[m] = acc;
}

// ---------------------------------------------------------------------------
extern "C" void kernel_launch(void* const* d_in, const int* in_sizes, int n_in,
                              void* d_out, int out_size, void* d_ws, size_t ws_size,
                              hipStream_t stream) {
  (void)in_sizes; (void)n_in; (void)out_size; (void)ws_size;
  const float* mesh  = (const float*)d_in[0];
  const float* cmap  = (const float*)d_in[1];
  const float* eps   = (const float*)d_in[2];
  const float* temb  = (const float*)d_in[3];
  const int*   toks  = (const int*)d_in[4];
  const int*   finit = (const int*)d_in[5];
  const float* pn_w1 = (const float*)d_in[6],  *pn_b1 = (const float*)d_in[7];
  const float* pn_w2 = (const float*)d_in[8],  *pn_b2 = (const float*)d_in[9];
  const float* pn_w3 = (const float*)d_in[10], *pn_b3 = (const float*)d_in[11];
  const float* ce_w1 = (const float*)d_in[12], *ce_b1 = (const float*)d_in[13];
  const float* ce_w2 = (const float*)d_in[14], *ce_b2 = (const float*)d_in[15];
  const float* ce_w3 = (const float*)d_in[16], *ce_b3 = (const float*)d_in[17];
  const float* mlp_w1= (const float*)d_in[18], *mlp_b1= (const float*)d_in[19];
  const float* mlp_w2= (const float*)d_in[20], *mlp_b2= (const float*)d_in[21];
  const float* mlp_w3= (const float*)d_in[22], *mlp_b3= (const float*)d_in[23];
  const float* dec_w1= (const float*)d_in[24], *dec_b1= (const float*)d_in[25];
  const float* dec_w2= (const float*)d_in[26], *dec_b2= (const float*)d_in[27];
  const float* dec_w3= (const float*)d_in[28], *dec_b3= (const float*)d_in[29];
  const float* dec_w4= (const float*)d_in[30], *dec_b4= (const float*)d_in[31];

  // output layout (flat, in reference return order)
  float* out    = (float*)d_out;
  float* refine = out;                 // [16,2048,1]
  float* samp   = out + 32768;         // [16,2048,1]
  float* gf     = out + 65536;         // [16,1024]
  float* sobj   = out + 81920;         // [16]
  float* tfeat  = out + 81936;         // [16,512]
  float* mu     = out + 90128;         // [16,64]
  float* lv     = out + 91152;         // [16,64]

  char* ws = (char*)d_ws;
  size_t cur = 0;
  auto alloc = [&](size_t bytes) -> char* {
    char* p = ws + cur; cur += (bytes + 255) & ~(size_t)255; return p;
  };
  int*      idx     = (int*)     alloc((size_t)MROWS * 4);
  float*    pc_norm = (float*)   alloc((size_t)MROWS * 3 * 4);
  float*    pcc     = (float*)   alloc((size_t)MROWS * 4 * 4);
  _Float16* h1p     = (_Float16*)alloc((size_t)MROWS * 64 * 2);   // pn local feat (persists)
  _Float16* h1c     = (_Float16*)alloc((size_t)MROWS * 64 * 2);
  _Float16* Bf      = (_Float16*)alloc((size_t)MROWS * 128 * 2);  // h2 (pn, ce), later d3
  _Float16* Cf      = (_Float16*)alloc((size_t)MROWS * 1024 * 2); // h3 (pn, ce), later d1+d2
  float*    g2      = (float*)   alloc(16 * 1024 * 4);
  float*    t1      = (float*)   alloc(16 * 512 * 4);
  float*    t2      = (float*)   alloc(16 * 256 * 4);
  float*    t3      = (float*)   alloc(16 * 128 * 4);
  float*    cv      = (float*)   alloc(16 * 64 * 4);
  float*    pb      = (float*)   alloc(16 * 512 * 4);
  _Float16* pnw2p   = (_Float16*)alloc(64  * 128  * 2);
  _Float16* pnw3p   = (_Float16*)alloc(128 * 1024 * 2);
  _Float16* cew2p   = (_Float16*)alloc(64  * 128  * 2);
  _Float16* cew3p   = (_Float16*)alloc(128 * 1024 * 2);
  _Float16* d1p     = (_Float16*)alloc(64  * 512  * 2);   // dec_w1 rows 1024..1087
  _Float16* d2p     = (_Float16*)alloc(512 * 256  * 2);
  _Float16* d3p     = (_Float16*)alloc(256 * 128  * 2);
  _Float16* d1 = Cf;                                 // [32768,512] reuse h3 space
  _Float16* d2 = Cf + (size_t)MROWS * 512;           // [32768,256]
  _Float16* d3 = Bf;                                 // [32768,128] reuse h2 space

  // geometry + sampling
  k_scale<<<BATCH, 256, 0, stream>>>(mesh, sobj);
  size_t fps_lds = (size_t)NPTS * 3 * 4 + 32 * 4 + 32 * 4 + 16;
  k_fps<<<BATCH, 1024, fps_lds, stream>>>(mesh, finit, idx);
  k_gather<<<MROWS/256, 256, 0, stream>>>(mesh, cmap, idx, sobj, pc_norm, pcc, samp);

  // pre-pack WMMA weight fragments (tiny, L2-resident)
  k_pack<<<(64*128  + 255)/256, 256, 0, stream>>>(pn_w2, 64, 128, 0, pnw2p);
  k_pack<<<(128*1024+ 255)/256, 256, 0, stream>>>(pn_w3, 128, 1024, 0, pnw3p);
  k_pack<<<(64*128  + 255)/256, 256, 0, stream>>>(ce_w2, 64, 128, 0, cew2p);
  k_pack<<<(128*1024+ 255)/256, 256, 0, stream>>>(ce_w3, 128, 1024, 0, cew3p);
  k_pack<<<(64*512  + 255)/256, 256, 0, stream>>>(dec_w1, 64, 512, 1024, d1p); // local_feat slice
  k_pack<<<(512*256 + 255)/256, 256, 0, stream>>>(dec_w2, 512, 256, 0, d2p);
  k_pack<<<(256*128 + 255)/256, 256, 0, stream>>>(dec_w3, 256, 128, 0, d3p);

  // PointNet (mesh)
  k_layer1<<<MROWS/4, 256, 0, stream>>>(pc_norm, 3, pn_w1, pn_b1, h1p);
  k_wmma<2, 1, false, true ><<<256, 256, 0, stream>>>(h1p, pnw2p, pn_b2, nullptr, Bf, 128);
  k_wmma<4, 1, false, true ><<<256, 256, 0, stream>>>(Bf, pnw3p, pn_b3, nullptr, Cf, 1024);
  k_maxpool<<<dim3(4, BATCH), 256, 0, stream>>>(Cf, 1024, gf);

  // Contact encoder
  k_layer1<<<MROWS/4, 256, 0, stream>>>(pcc, 4, ce_w1, ce_b1, h1c);
  k_wmma<2, 1, false, true ><<<256, 256, 0, stream>>>(h1c, cew2p, ce_b2, nullptr, Bf, 128);
  k_wmma<4, 1, false, true ><<<256, 256, 0, stream>>>(Bf, cew3p, ce_b3, nullptr, Cf, 1024);
  k_maxpool<<<dim3(4, BATCH), 256, 0, stream>>>(Cf, 1024, g2);

  // text + small MLP stack + reparameterize
  k_text<<<BATCH, 512, 0, stream>>>(temb, toks, tfeat);
  k_dense<<<BATCH, 512, 0, stream>>>(g2, mlp_w1, mlp_b1, t1, 1024, 512, 1);
  k_dense<<<BATCH, 256, 0, stream>>>(t1, mlp_w2, mlp_b2, t2, 512, 256, 1);
  k_dense<<<BATCH, 128, 0, stream>>>(t2, mlp_w3, mlp_b3, t3, 256, 128, 0);
  k_reparam<<<BATCH, 64, 0, stream>>>(t3, eps, mu, lv, cv);

  // decoder: fold all broadcast features into per-batch bias, WMMA the rest
  k_pbias<<<BATCH, 512, 0, stream>>>(gf, tfeat, sobj, cv, dec_w1, dec_b1, pb);
  k_wmma<2,  2, true,  false><<<256, 256, 0, stream>>>(h1p, d1p, nullptr, pb, d1, 512);
  k_wmma<16, 2, false, true ><<<256, 256, 0, stream>>>(d1, d2p, dec_b2, nullptr, d2, 256);
  k_wmma<8,  2, false, true ><<<256, 256, 0, stream>>>(d2, d3p, dec_b3, nullptr, d3, 128);
  k_dec4<<<MROWS/256, 256, 0, stream>>>(d3, dec_w4, dec_b4, refine);
}
